// PartitionAttention_81320910783025
// MI455X (gfx1250) — compile-verified, matrix operand order checked by reading, not run on hardware
//
#include <hip/hip_runtime.h>

typedef _Float16 h16;
typedef __attribute__((ext_vector_type(16))) _Float16 v16h;
typedef __attribute__((ext_vector_type(8)))  _Float16 v8h;
typedef __attribute__((ext_vector_type(8)))  float    v8f;

#define C_      256
#define HW_     256
#define HEADS   16
#define NW      64          // tokens per window
#define NWINDOWS 2048       // 2 * 32 * 32
#define QKVW_ELEMS  196608  // 256*768  (= 48*8 fragments * 512 elems)
#define PROJW_ELEMS 65536   // 256*256  (= 16*8 fragments * 512 elems)
#define BIAS_ELEMS  65536   // 16*64*64
#define EPS_LN 1e-5f

// ---------------- WMMA helpers (CDNA5 16x16x32 f16 -> f32) ----------------

__device__ __forceinline__ v8f wmma_f16(v16h a, v16h b, v8f c) {
  // 8 args: (neg_a, A, neg_b, B, c_mod, C, reuse_a, reuse_b)
  return __builtin_amdgcn_wmma_f32_16x16x32_f16(false, a, false, b, (short)0, c,
                                                false, false);
}

// Per-lane K-index pattern for 16-bit A/B fragments (ISA 7.12.2):
// lanes 0-15 hold K {0..7,16..23}, lanes 16-31 hold K {8..15,24..31}.
__device__ __forceinline__ int kb_of(int v, int lane) {
  return ((v >> 2) << 4) + ((lane >> 4) << 3) + ((v & 3) << 1);
}

// A-matrix 16x32 f16 loader from row-major storage (row stride ld elems).
__device__ __forceinline__ v16h load_a(const h16* p, int ld, int m0, int k0,
                                       int lane, int kvalid) {
  const int m = m0 + (lane & 15);
  v16h a;
#pragma unroll
  for (int v = 0; v < 8; ++v) {
    const int k = k0 + kb_of(v, lane);
    a[2 * v]     = (k     < kvalid) ? p[m * ld + k]     : (h16)0.0f;
    a[2 * v + 1] = (k + 1 < kvalid) ? p[m * ld + k + 1] : (h16)0.0f;
  }
  return a;
}

// B-matrix loader where storage holds the transpose: element (k,n) = p[(n0+n)*ld + k0+k]
__device__ __forceinline__ v16h load_b_tr(const h16* p, int ld, int k0, int n0,
                                          int lane, int kvalid) {
  const int n = n0 + (lane & 15);
  v16h b;
#pragma unroll
  for (int v = 0; v < 8; ++v) {
    const int k = k0 + kb_of(v, lane);
    b[2 * v]     = (k     < kvalid) ? p[n * ld + k]     : (h16)0.0f;
    b[2 * v + 1] = (k + 1 < kvalid) ? p[n * ld + k + 1] : (h16)0.0f;
  }
  return b;
}

// Pre-swizzled B fragment: one contiguous 32B chunk per lane.
__device__ __forceinline__ v16h load_b_frag(const h16* base, int frag, int lane) {
  return *(const v16h*)(base + (frag << 9) + (lane << 4));
}

// ---------------- prep: swizzle weights into B-fragment order + bias gather ----

__global__ void prep_kernel(const float* __restrict__ qkvw,
                            const float* __restrict__ projw,
                            const float* __restrict__ table,
                            const int*   __restrict__ relidx,
                            h16* __restrict__ qkvwh,
                            h16* __restrict__ projwh,
                            float* __restrict__ biasg) {
  const int i = blockIdx.x * 256 + threadIdx.x;
  if (i < QKVW_ELEMS) {
    const int f = i >> 9, l = (i >> 4) & 31, e = i & 15;
    const int nt = f >> 3, kt = f & 7;
    const int n = nt * 16 + (l & 15);
    const int k = kt * 32 + kb_of(e >> 1, l) + (e & 1);
    qkvwh[i] = (h16)qkvw[k * 768 + n];
  }
  const int j = i - QKVW_ELEMS;
  if (j >= 0 && j < PROJW_ELEMS) {
    const int f = j >> 9, l = (j >> 4) & 31, e = j & 15;
    const int nt = f >> 3, kt = f & 7;
    const int n = nt * 16 + (l & 15);
    const int k = kt * 32 + kb_of(e >> 1, l) + (e & 1);
    projwh[j] = (h16)projw[k * 256 + n];
  }
  const int k2 = i - (QKVW_ELEMS + PROJW_ELEMS);
  if (k2 >= 0 && k2 < BIAS_ELEMS) {
    const int h   = k2 >> 12;
    const int rem = k2 & 4095;
    const int ii  = rem >> 6;
    const int jj  = rem & 63;
    biasg[k2] = table[relidx[ii * 64 + jj] * HEADS + h];
  }
}

// ---------------- main: one workgroup per 8x8 window ----------------
// Dynamic LDS layout (bytes):
//   xw   f32 [64][256]        @ 0       (65536)
//   xn   f16 [64][256]        @ 65536   (32768)
//   qn   f16 [16][64][16]     @ 98304   (32768)   q, [head][token][hd]
//   kn   f16 [16][64][16]     @ 131072  (32768)   k, [head][token][hd]
//   vt   f16 [16][16][64]     @ 163840  (32768)   v TRANSPOSED, [head][hd][token]
//   pbuf f16 [8 waves][16][64]@ 196608  (16384)
//   ao   f16 [64][256]        @ 212992  (32768)
//   stats f32 [64][2]         @ 245760  (512)
#define SMEM_BYTES 246272

__global__ void window_attn_kernel(const float* __restrict__ X,
                                   const float* __restrict__ pre_g,
                                   const float* __restrict__ pre_b,
                                   const float* __restrict__ post_g,
                                   const float* __restrict__ post_b,
                                   const float* __restrict__ qkv_b,
                                   const float* __restrict__ proj_b,
                                   const h16*   __restrict__ qkvwh,   // swizzled
                                   const h16*   __restrict__ projwh,  // swizzled
                                   const float* __restrict__ biasg,   // [16][64][64]
                                   float* __restrict__ out) {
  extern __shared__ char smem[];
  float* xw    = (float*)(smem);
  h16*   xn    = (h16*)(smem + 65536);
  h16*   qn    = (h16*)(smem + 98304);
  h16*   kn    = (h16*)(smem + 131072);
  h16*   vt    = (h16*)(smem + 163840);
  h16*   pbuf  = (h16*)(smem + 196608);
  h16*   ao    = (h16*)(smem + 212992);
  float* stats = (float*)(smem + 245760);

  const int tid  = threadIdx.x;
  const int lane = tid & 31;
  const int wave = __builtin_amdgcn_readfirstlane(tid >> 5);  // SGPR-uniform

  const int wi  = blockIdx.x;
  const int wwi = wi & 31;
  const int hwi = (wi >> 5) & 31;
  const int bb  = wi >> 10;

  // ---- Phase 1: load window -> LDS. 4 consecutive channels per thread so the
  //      transposed LDS writes are b128 (16B) instead of scalar b32. ----
#pragma unroll
  for (int it = 0; it < 2; ++it) {
    const int g  = it * 256 + tid;      // 512 groups: (c-quad, i)
    const int c0 = (g >> 3) << 2;
    const int i  = g & 7;
    float rv[4][8];
#pragma unroll
    for (int r = 0; r < 4; ++r) {
      const float* gp =
          X + (((size_t)(bb * C_ + c0 + r)) * HW_ + hwi * 8 + i) * HW_ + wwi * 8;
      const float4 p0 = *(const float4*)gp;
      const float4 p1 = *(const float4*)(gp + 4);
      rv[r][0] = p0.x; rv[r][1] = p0.y; rv[r][2] = p0.z; rv[r][3] = p0.w;
      rv[r][4] = p1.x; rv[r][5] = p1.y; rv[r][6] = p1.z; rv[r][7] = p1.w;
    }
#pragma unroll
    for (int j = 0; j < 8; ++j) {
      *(float4*)(xw + (i * 8 + j) * C_ + c0) =
          make_float4(rv[0][j], rv[1][j], rv[2][j], rv[3][j]);
    }
  }
  __syncthreads();

  // ---- Phase 2: pre-LayerNorm, 4 threads/row, b128 reads, fused sum/sumsq ----
  {
    const int row  = tid >> 2;
    const int part = tid & 3;
    const float* rp = xw + row * C_ + part * 64;
    float s = 0.f, s2 = 0.f;
#pragma unroll
    for (int q = 0; q < 16; ++q) {
      const float4 x4 = *(const float4*)(rp + q * 4);
      s  += x4.x + x4.y + x4.z + x4.w;
      s2 += x4.x * x4.x + x4.y * x4.y + x4.z * x4.z + x4.w * x4.w;
    }
    s  += __shfl_xor(s, 1, 4);  s  += __shfl_xor(s, 2, 4);
    s2 += __shfl_xor(s2, 1, 4); s2 += __shfl_xor(s2, 2, 4);
    const float mean = s * (1.0f / C_);
    const float var  = s2 * (1.0f / C_) - mean * mean;
    const float inv  = rsqrtf(var + EPS_LN);
    h16* xrow = xn + row * C_ + part * 64;
    const float* gp = pre_g + part * 64;
    const float* bp = pre_b + part * 64;
#pragma unroll
    for (int q = 0; q < 8; ++q) {  // 8 halves per packed 16B store
      const float4 xa = *(const float4*)(rp + q * 8);
      const float4 xb = *(const float4*)(rp + q * 8 + 4);
      const float4 ga = *(const float4*)(gp + q * 8);
      const float4 gb = *(const float4*)(gp + q * 8 + 4);
      const float4 ba = *(const float4*)(bp + q * 8);
      const float4 b2 = *(const float4*)(bp + q * 8 + 4);
      v8h pk;
      pk[0] = (h16)((xa.x - mean) * inv * ga.x + ba.x);
      pk[1] = (h16)((xa.y - mean) * inv * ga.y + ba.y);
      pk[2] = (h16)((xa.z - mean) * inv * ga.z + ba.z);
      pk[3] = (h16)((xa.w - mean) * inv * ga.w + ba.w);
      pk[4] = (h16)((xb.x - mean) * inv * gb.x + b2.x);
      pk[5] = (h16)((xb.y - mean) * inv * gb.y + b2.y);
      pk[6] = (h16)((xb.z - mean) * inv * gb.z + b2.z);
      pk[7] = (h16)((xb.w - mean) * inv * gb.w + b2.w);
      *(v8h*)(xrow + q * 8) = pk;
    }
  }
  __syncthreads();

  // ---- Phase 3: QKV GEMM [64,256]x[256,768], in-register l2norm of q,k ----
  for (int tile = wave; tile < 192; tile += 8) {
    const int mt = tile / 48;
    const int nt = tile % 48;
    {  // prefetch next tile's weight fragments (L2 -> near caches)
      const int tile2 = tile + 8;
      if (tile2 < 192)
        __builtin_prefetch(qkvwh + (((tile2 % 48) * 8) << 9) + (lane << 4), 0, 1);
    }
    v8f acc = {};
#pragma unroll
    for (int kt = 0; kt < 8; ++kt) {
      const v16h a = load_a(xn, C_, mt * 16, kt * 32, lane, C_);
      const v16h b = load_b_frag(qkvwh, nt * 8 + kt, lane);
      acc = wmma_f16(a, b, acc);
    }
    const float badd = qkv_b[nt * 16 + (lane & 15)];
#pragma unroll
    for (int r = 0; r < 8; ++r) acc[r] += badd;

    if (nt < 32) {  // q or k: l2norm over hd=16 (= N tile) per row, store [tok][hd]
#pragma unroll
      for (int r = 0; r < 8; ++r) {
        float s = acc[r] * acc[r];
        s += __shfl_xor(s, 1, 16);
        s += __shfl_xor(s, 2, 16);
        s += __shfl_xor(s, 4, 16);
        s += __shfl_xor(s, 8, 16);
        const float inv = 1.0f / fmaxf(sqrtf(s), 1e-12f);
        acc[r] *= inv;
      }
      h16* dst = (nt < 16) ? (qn + nt * (NW * 16)) : (kn + (nt - 16) * (NW * 16));
#pragma unroll
      for (int r = 0; r < 8; ++r) {
        const int m   = r + ((lane >> 4) << 3);
        const int tok = mt * 16 + m;
        dst[tok * 16 + (lane & 15)] = (h16)acc[r];
      }
    } else {  // v: store TRANSPOSED [hd][token]; 8 rows pack to one b128 store
      h16* vh = vt + (nt - 32) * (NW * 16);
      v8h pk;
#pragma unroll
      for (int r = 0; r < 8; ++r) pk[r] = (h16)acc[r];
      *(v8h*)(vh + (lane & 15) * NW + mt * 16 + ((lane >> 4) << 3)) = pk;
    }
  }
  __syncthreads();

  // ---- Phase 4: per-(head, row-strip) attention, softmax in registers ----
  for (int strip = wave; strip < 64; strip += 8) {
    const int h  = strip >> 2;
    const int mt = strip & 3;
    const h16* qh = qn + h * (NW * 16);
    const h16* kh = kn + h * (NW * 16);
    const h16* vh = vt + h * (NW * 16);
    const float* bh = biasg + h * (NW * NW);

    v8f s[4];
    const v16h aq = load_a(qh, 16, mt * 16, 0, lane, 16);  // K padded 16->32
#pragma unroll
    for (int nt = 0; nt < 4; ++nt) {
      const v16h bk = load_b_tr(kh, 16, 0, nt * 16, lane, 16);
      v8f z = {};
      s[nt] = wmma_f16(aq, bk, z);
    }
    // + relative position bias
#pragma unroll
    for (int nt = 0; nt < 4; ++nt)
#pragma unroll
      for (int r = 0; r < 8; ++r) {
        const int irow = mt * 16 + r + ((lane >> 4) << 3);
        const int j    = nt * 16 + (lane & 15);
        s[nt][r] += bh[irow * NW + j];
      }
    // softmax over 64 cols (4 tiles x 16 lanes)
#pragma unroll
    for (int r = 0; r < 8; ++r) {
      float mx = fmaxf(fmaxf(s[0][r], s[1][r]), fmaxf(s[2][r], s[3][r]));
      mx = fmaxf(mx, __shfl_xor(mx, 1, 16));
      mx = fmaxf(mx, __shfl_xor(mx, 2, 16));
      mx = fmaxf(mx, __shfl_xor(mx, 4, 16));
      mx = fmaxf(mx, __shfl_xor(mx, 8, 16));
      float sum = 0.f;
#pragma unroll
      for (int nt = 0; nt < 4; ++nt) {
        const float e = __expf(s[nt][r] - mx);
        s[nt][r] = e;
        sum += e;
      }
      sum += __shfl_xor(sum, 1, 16);
      sum += __shfl_xor(sum, 2, 16);
      sum += __shfl_xor(sum, 4, 16);
      sum += __shfl_xor(sum, 8, 16);
      const float inv = 1.0f / sum;
#pragma unroll
      for (int nt = 0; nt < 4; ++nt) s[nt][r] *= inv;
    }
    // bounce P through per-wave LDS to re-layout D->A (wave-private, no barrier)
    h16* pw = pbuf + wave * (16 * 64);
#pragma unroll
    for (int nt = 0; nt < 4; ++nt)
#pragma unroll
      for (int r = 0; r < 8; ++r) {
        const int m = r + ((lane >> 4) << 3);
        const int j = nt * 16 + (lane & 15);
        pw[m * 64 + j] = (h16)s[nt][r];
      }
    asm volatile("s_wait_dscnt 0x0" ::: "memory");
    // O = P[16x64] x V[64x16]; V transposed -> contiguous LDS b128 runs
    v8f o = {};
#pragma unroll
    for (int kk = 0; kk < 2; ++kk) {
      const v16h ap = load_a(pw, 64, 0, kk * 32, lane, 64);
      const v16h bv = load_b_tr(vh, 64, kk * 32, 0, lane, 64);
      o = wmma_f16(ap, bv, o);
    }
#pragma unroll
    for (int r = 0; r < 8; ++r) {
      const int m   = r + ((lane >> 4) << 3);
      const int tok = mt * 16 + m;
      ao[tok * C_ + h * 16 + (lane & 15)] = (h16)o[r];
    }
  }
  __syncthreads();

  // ---- Phase 5: proj GEMM [64,256]x[256,256] + residual into xw ----
  for (int tile = wave; tile < 64; tile += 8) {
    const int mt = tile >> 4;
    const int nt = tile & 15;
    {
      const int tile2 = tile + 8;
      if (tile2 < 64)
        __builtin_prefetch(projwh + (((tile2 & 15) * 8) << 9) + (lane << 4), 0, 1);
    }
    v8f acc = {};
#pragma unroll
    for (int kt = 0; kt < 8; ++kt) {
      const v16h a = load_a(ao, C_, mt * 16, kt * 32, lane, C_);
      const v16h b = load_b_frag(projwh, nt * 8 + kt, lane);
      acc = wmma_f16(a, b, acc);
    }
    const float badd = proj_b[nt * 16 + (lane & 15)];
#pragma unroll
    for (int r = 0; r < 8; ++r) {
      const int m   = r + ((lane >> 4) << 3);
      const int tok = mt * 16 + m;
      const int c   = nt * 16 + (lane & 15);
      xw[tok * C_ + c] += acc[r] + badd;   // residual (single writer per elem)
    }
  }
  __syncthreads();

  // ---- Phase 6: post-LN statistics, 4 threads/row with b128 reads ----
  {
    const int row  = tid >> 2;
    const int part = tid & 3;
    const float* rp = xw + row * C_ + part * 64;
    float s = 0.f, s2 = 0.f;
#pragma unroll
    for (int q = 0; q < 16; ++q) {
      const float4 x4 = *(const float4*)(rp + q * 4);
      s  += x4.x + x4.y + x4.z + x4.w;
      s2 += x4.x * x4.x + x4.y * x4.y + x4.z * x4.z + x4.w * x4.w;
    }
    s  += __shfl_xor(s, 1, 4);  s  += __shfl_xor(s, 2, 4);
    s2 += __shfl_xor(s2, 1, 4); s2 += __shfl_xor(s2, 2, 4);
    if (part == 0) {
      const float mean = s * (1.0f / C_);
      const float var  = s2 * (1.0f / C_) - mean * mean;
      stats[row * 2 + 0] = mean;
      stats[row * 2 + 1] = rsqrtf(var + EPS_LN);
    }
  }
  __syncthreads();

  // ---- Phase 7: x + LN(x), b128 LDS reads (4-channel groups), coalesced store --
#pragma unroll
  for (int it = 0; it < 2; ++it) {
    const int g  = it * 256 + tid;
    const int c0 = (g >> 3) << 2;
    const int i  = g & 7;
    const float4 gv = *(const float4*)(post_g + c0);
    const float4 bv = *(const float4*)(post_b + c0);
    float ov[4][8];
#pragma unroll
    for (int j = 0; j < 8; ++j) {
      const int t = i * 8 + j;
      const float4 x4 = *(const float4*)(xw + t * C_ + c0);
      const float m  = stats[t * 2 + 0];
      const float iv = stats[t * 2 + 1];
      ov[0][j] = x4.x + (x4.x - m) * iv * gv.x + bv.x;
      ov[1][j] = x4.y + (x4.y - m) * iv * gv.y + bv.y;
      ov[2][j] = x4.z + (x4.z - m) * iv * gv.z + bv.z;
      ov[3][j] = x4.w + (x4.w - m) * iv * gv.w + bv.w;
    }
#pragma unroll
    for (int r = 0; r < 4; ++r) {
      float* op =
          out + (((size_t)(bb * C_ + c0 + r)) * HW_ + hwi * 8 + i) * HW_ + wwi * 8;
      *(float4*)op       = make_float4(ov[r][0], ov[r][1], ov[r][2], ov[r][3]);
      *(float4*)(op + 4) = make_float4(ov[r][4], ov[r][5], ov[r][6], ov[r][7]);
    }
  }
}

// ---------------- host-side launch ----------------

extern "C" void kernel_launch(void* const* d_in, const int* in_sizes, int n_in,
                              void* d_out, int out_size, void* d_ws, size_t ws_size,
                              hipStream_t stream) {
  const float* X      = (const float*)d_in[0];
  const float* pre_g  = (const float*)d_in[1];
  const float* pre_b  = (const float*)d_in[2];
  const float* post_g = (const float*)d_in[3];
  const float* post_b = (const float*)d_in[4];
  const float* qkv_w  = (const float*)d_in[5];
  const float* qkv_b  = (const float*)d_in[6];
  const float* proj_w = (const float*)d_in[7];
  const float* proj_b = (const float*)d_in[8];
  const float* table  = (const float*)d_in[9];
  const int*   relidx = (const int*)d_in[10];
  float* out = (float*)d_out;

  char* ws = (char*)d_ws;
  h16*   qkvwh  = (h16*)ws;                                  // 393216 B
  h16*   projwh = (h16*)(ws + 393216);                       // 131072 B
  float* biasg  = (float*)(ws + 393216 + 131072);            // 262144 B

  (void)hipFuncSetAttribute((const void*)window_attn_kernel,
                            hipFuncAttributeMaxDynamicSharedMemorySize,
                            SMEM_BYTES);

  const int prep_total = QKVW_ELEMS + PROJW_ELEMS + BIAS_ELEMS;
  prep_kernel<<<(prep_total + 255) / 256, 256, 0, stream>>>(
      qkv_w, proj_w, table, relidx, qkvwh, projwh, biasg);

  window_attn_kernel<<<NWINDOWS, 256, SMEM_BYTES, stream>>>(
      X, pre_g, pre_b, post_g, post_b, qkv_b, proj_b, qkvwh, projwh, biasg, out);
}
